// Model_45140106281349
// MI455X (gfx1250) — matrix operand order repaired and checked by reading, not compiled, not run on hardware
//
#include <hip/hip_runtime.h>
#include <math.h>

// ---------------------------------------------------------------------------
// CARD-style dual-attention transformer forward for MI455X (gfx1250).
// Large GEMMs: v_wmma_f32_16x16x32_f16, 64x128 block tile, 2x2 WMMA tiles per
// wave, double-buffered Tensor-Data-Mover DMA of A/B tiles overlapped with
// compute; LDS tiles held in fragment order (2x ds_load_b128 per fragment).
// ---------------------------------------------------------------------------

typedef __attribute__((ext_vector_type(16))) _Float16 v16h;
typedef __attribute__((ext_vector_type(8)))  _Float16 v8h;
typedef __attribute__((ext_vector_type(8)))  float    v8f;
typedef __attribute__((ext_vector_type(4)))  float    f32x4;

#if defined(__AMDGCN__) && __has_builtin(__builtin_amdgcn_tensor_load_to_lds)
#define USE_TDM 1
#else
#define USE_TDM 0
#endif

#if USE_TDM
typedef __attribute__((ext_vector_type(4))) unsigned int u32x4;
typedef __attribute__((ext_vector_type(8))) int          i32x8;
typedef __attribute__((ext_vector_type(4))) int          i32x4;

// Issue a 2D TENSOR_LOAD_TO_LDS: tile (tile_d1 rows x tile_d0 elems) of a
// row-major f32 tensor (row stride d0_stride elems) -> contiguous LDS tile.
// Rows/cols beyond (tensor_d1, tensor_d0) are zero-filled by the TDM.
// D# bitfields per CDNA5 ISA 8.3/8.4 (group0 128b, group1 256b).
__device__ __forceinline__ void tdm_load_2d(const float* gptr, unsigned lds_off,
                                            unsigned tensor_d0, unsigned tensor_d1,
                                            unsigned tile_d0, unsigned tile_d1,
                                            unsigned long long d0_stride)
{
    unsigned long long ga = (unsigned long long)(size_t)gptr;
    u32x4 g0;
    g0[0] = 1u;                                            // count=1 (valid user D#)
    g0[1] = lds_off;                                       // lds_addr
    g0[2] = (unsigned)(ga & 0xffffffffu);                  // global_addr[31:0]
    g0[3] = (unsigned)((ga >> 32) & 0x1ffffffu) | (2u << 30); // addr[56:32] | type=2
    i32x8 g1;
    g1[0] = (int)(2u << 16);                               // data_size=4B, wg_mask=0
    g1[1] = (int)((tensor_d0 & 0xffffu) << 16);            // tensor_dim0[15:0]
    g1[2] = (int)(((tensor_d0 >> 16) & 0xffffu) | ((tensor_d1 & 0xffffu) << 16));
    g1[3] = (int)(((tensor_d1 >> 16) & 0xffffu) | ((tile_d0 & 0xffffu) << 16));
    g1[4] = (int)(tile_d1 & 0xffffu);                      // tile_dim1 (tile_dim2=0)
    g1[5] = (int)(d0_stride & 0xffffffffu);                // dim0_stride[31:0]
    g1[6] = (int)((d0_stride >> 32) & 0xffffu);            // dim0_stride[47:32]
    g1[7] = 0;
    i32x4 z4 = {0, 0, 0, 0};
#if defined(__clang_major__) && (__clang_major__ >= 23)
    i32x8 z8 = {0, 0, 0, 0, 0, 0, 0, 0};
    __builtin_amdgcn_tensor_load_to_lds(g0, g1, z4, z4, z8, 0);
#else
    __builtin_amdgcn_tensor_load_to_lds(g0, g1, z4, z4, 0);
#endif
}
#endif  // USE_TDM

__device__ __forceinline__ float gelu_exact(float x) {
    return 0.5f * x * (1.0f + erff(x * 0.70710678118654752f));
}

// EMA matrix element (alpha = 0.3); pw[i] = 0.7^i precomputed in LDS.
__device__ __forceinline__ float emaE(int i, int j, const float* pw) {
    if (i == j) return (i == 0) ? 1.0f : 0.3f;
    if (j < i)  return (j == 0) ? pw[i] : 0.3f * pw[i - j];
    return 0.0f;
}

// A-fragment LDS slot for K index k (ISA 7.12.2 16-bit A layout):
// lane half h reads contiguous halves [h*16, h*16+16); slot(k) puts A[m][k]
// where that lane expects it.
__device__ __forceinline__ int slotA(int k) {
    return (((k >> 3) & 1) << 4) + (((k >> 4) & 1) << 3) + (k & 7);
}

// 2x2 WMMA tiles per wave from fragment-ordered LDS (2x ds_load_b128 each).
__device__ __forceinline__ void wmma_tile(const _Float16 (*As2)[40],
                                          const _Float16 (*Bs2)[40],
                                          int wm, int wn, int half, int l16,
                                          v8f acc[2][2])
{
    v16h af[2], bf[2];
#pragma unroll
    for (int mt = 0; mt < 2; ++mt) {
        const _Float16* p = &As2[wm + mt * 16 + l16][half * 16];
        v8h lo = *(const v8h*)p;
        v8h hi = *(const v8h*)(p + 8);
        af[mt] = __builtin_shufflevector(lo, hi, 0, 1, 2, 3, 4, 5, 6, 7,
                                         8, 9, 10, 11, 12, 13, 14, 15);
    }
#pragma unroll
    for (int nt = 0; nt < 2; ++nt) {
        const _Float16* p = &Bs2[wn + nt * 16 + l16][half * 16];
        v8h lo = *(const v8h*)p;
        v8h hi = *(const v8h*)(p + 8);
        bf[nt] = __builtin_shufflevector(lo, hi, 0, 1, 2, 3, 4, 5, 6, 7,
                                         8, 9, 10, 11, 12, 13, 14, 15);
    }
#pragma unroll
    for (int mt = 0; mt < 2; ++mt)
#pragma unroll
        for (int nt = 0; nt < 2; ++nt)
            acc[mt][nt] = __builtin_amdgcn_wmma_f32_16x16x32_f16(
                false, af[mt], false, bf[nt], (short)0, acc[mt][nt], false, false);
}

// ---------------------------------------------------------------------------
// WMMA GEMM: Y[M][N] = act(X[M][K] @ W[N][K]^T + bias[N])
// 256 threads = 8 waves; block tile 64(M) x 128(N); wave tile 32x32 (2x2 WMMA).
// Requires K%32==0 (all call sites). guard==1 only masks ragged output stores;
// staging OOB is zero-filled (by the TDM descriptor, or row guards in the
// fallback path).
// ---------------------------------------------------------------------------
template <int ACT>
__global__ __launch_bounds__(256)
void gemm_wmma(const float* __restrict__ X, const float* __restrict__ W,
               const float* __restrict__ bias, float* __restrict__ Y,
               int M, int N, int K, int guard)
{
    __shared__ __align__(16) _Float16 As2[64][40];    // [m][frag-slot]
    __shared__ __align__(16) _Float16 Bs2[128][40];   // [n][frag-slot]
#if USE_TDM
    __shared__ __align__(16) float stA[2][64 * 32];   // double-buffered DMA staging
    __shared__ __align__(16) float stB[2][128 * 32];
#endif

    const int tid  = threadIdx.x;
    const int lane = tid & 31;
    const int wave = tid >> 5;
    const int half = lane >> 4;
    const int l16  = lane & 15;

    const int bm = blockIdx.y * 64;
    const int bn = blockIdx.x * 128;
    const int wm = (wave >> 2) * 32;   // 0 or 32
    const int wn = (wave & 3) * 32;    // 0,32,64,96

    v8f acc[2][2];
#pragma unroll
    for (int i = 0; i < 2; ++i)
#pragma unroll
        for (int j = 0; j < 2; ++j)
            acc[i][j] = (v8f){0.f, 0.f, 0.f, 0.f, 0.f, 0.f, 0.f, 0.f};

    const int ntiles = K >> 5;

#if USE_TDM
    // -------- double-buffered TDM pipeline: DMA tile t+1 while computing t --
    if (wave == 0) {
        tdm_load_2d(X + (size_t)bm * K, (unsigned)(size_t)&stA[0][0],
                    (unsigned)K, (unsigned)M, 32u, 64u, (unsigned long long)K);
        tdm_load_2d(W + (size_t)bn * K, (unsigned)(size_t)&stB[0][0],
                    (unsigned)K, (unsigned)N, 32u, 128u, (unsigned long long)K);
    }
    for (int t = 0; t < ntiles; ++t) {
        const int cur = t & 1;
        if (wave == 0) {
            if (t + 1 < ntiles) {
                const int k0 = (t + 1) << 5;
                tdm_load_2d(X + (size_t)bm * K + k0, (unsigned)(size_t)&stA[cur ^ 1][0],
                            (unsigned)K, (unsigned)M, 32u, 64u, (unsigned long long)K);
                tdm_load_2d(W + (size_t)bn * K + k0, (unsigned)(size_t)&stB[cur ^ 1][0],
                            (unsigned)K, (unsigned)N, 32u, 128u, (unsigned long long)K);
                // Tile t's two loads precede the two just issued; TDM completes
                // in order per wave, so <=2 outstanding => tile t has landed.
                __builtin_amdgcn_s_wait_tensorcnt(2);
            } else {
                __builtin_amdgcn_s_wait_tensorcnt(0);
            }
        }
        __syncthreads();
        asm volatile("" ::: "memory");
        // Convert f32 staging -> fragment-ordered f16 tiles.
        for (int i = tid; i < 64 * 32; i += 256) {
            int r = i >> 5, k = i & 31;
            As2[r][slotA(k)] = (_Float16)stA[cur][i];
        }
        for (int i = tid; i < 128 * 32; i += 256) {
            int r = i >> 5, k = i & 31;
            Bs2[r][slotA(k)] = (_Float16)stB[cur][i];
        }
        __syncthreads();
        wmma_tile(As2, Bs2, wm, wn, half, l16, acc);
        __syncthreads();
    }
#else
    // -------- fallback: vectorized float4 staging with row guards ----------
    for (int t = 0; t < ntiles; ++t) {
        const int k0 = t << 5;
        for (int q = tid; q < 64 * 8; q += 256) {
            int r = q >> 3, k4 = (q & 7) * 4;
            int gr = bm + r;
            f32x4 v = (gr < M) ? *(const f32x4*)&X[(size_t)gr * K + k0 + k4]
                               : (f32x4){0.f, 0.f, 0.f, 0.f};
            int s0 = slotA(k4);
#pragma unroll
            for (int j = 0; j < 4; ++j) As2[r][s0 + j] = (_Float16)v[j];
        }
        for (int q = tid; q < 128 * 8; q += 256) {
            int r = q >> 3, k4 = (q & 7) * 4;
            int gn = bn + r;
            f32x4 v = (gn < N) ? *(const f32x4*)&W[(size_t)gn * K + k0 + k4]
                               : (f32x4){0.f, 0.f, 0.f, 0.f};
#pragma unroll
            for (int j = 0; j < 4; ++j) Bs2[r][slotA(k4) + j] = (_Float16)v[j];
        }
        if (k0 + 32 < K) {
            __builtin_prefetch(&X[(size_t)(bm + (tid & 63)) * K + k0 + 32], 0, 2);
            __builtin_prefetch(&W[(size_t)(bn + (tid & 127)) * K + k0 + 32], 0, 2);
        }
        __syncthreads();
        wmma_tile(As2, Bs2, wm, wn, half, l16, acc);
        __syncthreads();
    }
#endif

    // Store: lane -> col l16; VGPR r -> row r + half*8.
#pragma unroll
    for (int mt = 0; mt < 2; ++mt) {
#pragma unroll
        for (int nt = 0; nt < 2; ++nt) {
            int col = bn + wn + nt * 16 + l16;
            if (guard && col >= N) continue;
            float bv = bias ? bias[col] : 0.0f;
#pragma unroll
            for (int r = 0; r < 8; ++r) {
                int row = bm + wm + mt * 16 + r + half * 8;
                if (guard && row >= M) continue;
                float v = acc[mt][nt][r] + bv;
                if (ACT == 1) v = gelu_exact(v);
                Y[(size_t)row * N + col] = v;
            }
        }
    }
}

// ---------------------------------------------------------------------------
// Per-(b,c) mean / unbiased-std over S=720 of x (B,S,CIN)
// ---------------------------------------------------------------------------
__global__ __launch_bounds__(256)
void rowstats_kernel(const float* __restrict__ x,
                     float* __restrict__ zm, float* __restrict__ zs)
{
    int bc = blockIdx.x;
    int b = bc >> 6, c = bc & 63;
    float s = 0.f, s2 = 0.f;
    for (int t = threadIdx.x; t < 720; t += 256) {
        float v = x[((size_t)b * 720 + t) * 64 + c];
        s += v; s2 += v * v;
    }
    __shared__ float sh[256], sh2[256];
    sh[threadIdx.x] = s; sh2[threadIdx.x] = s2;
    __syncthreads();
    for (int o = 128; o > 0; o >>= 1) {
        if (threadIdx.x < o) { sh[threadIdx.x] += sh[threadIdx.x + o];
                               sh2[threadIdx.x] += sh2[threadIdx.x + o]; }
        __syncthreads();
    }
    if (threadIdx.x == 0) {
        float m = sh[0] / 720.0f;
        float var = (sh2[0] - 720.0f * m * m) / 719.0f;
        zm[bc] = m;
        zs[bc] = sqrtf(fmaxf(var, 0.0f));
    }
}

// ---------------------------------------------------------------------------
// Normalize + unfold + patch-embed + pos + cls  -> inputs (B,64,90,256)
// ---------------------------------------------------------------------------
__global__ void embed_kernel(const float* __restrict__ x,
                             const float* __restrict__ wip_w,
                             const float* __restrict__ wip_b,
                             const float* __restrict__ pos,
                             const float* __restrict__ cls,
                             const float* __restrict__ zm,
                             const float* __restrict__ zs,
                             float* __restrict__ inputs)
{
    const size_t total = (size_t)16 * 64 * 90 * 256;
    size_t i = (size_t)blockIdx.x * 256 + threadIdx.x;
    if (i >= total) return;
    int d = (int)(i & 255);
    size_t j = i >> 8;
    int t = (int)(j % 90); j /= 90;
    int c = (int)(j % 64);
    int b = (int)(j / 64);
    float val;
    if (t == 0) {
        val = cls[d];
    } else {
        int p = t - 1;
        int bc = b * 64 + c;
        float m = zm[bc], sd = zs[bc] + 1e-4f;
        float s = wip_b[d];
#pragma unroll
        for (int q = 0; q < 16; ++q) {
            int srow = p * 8 + q;
            float zn = (x[((size_t)b * 720 + srow) * 64 + c] - m) / sd;
            s += zn * wip_w[d * 16 + q];
        }
        val = s + pos[p * 256 + d];
    }
    inputs[i] = val;
}

// dst[b][y][x][d] = src[b][x][y][d]   (B=16, D=256 fixed)
__global__ void transpose_ct_kernel(const float* __restrict__ src,
                                    float* __restrict__ dst, int X, int Y)
{
    size_t total = (size_t)16 * X * Y * 256;
    size_t i = (size_t)blockIdx.x * 256 + threadIdx.x;
    if (i >= total) return;
    int d = (int)(i & 255);
    size_t j = i >> 8;
    int xx = (int)(j % X); j /= X;
    int yy = (int)(j % Y);
    int b  = (int)(j / Y);
    dst[i] = src[(((size_t)b * X + xx) * Y + yy) * 256 + d];
}

// ---------------------------------------------------------------------------
// Over-hidden attention (oc): one block per (b, n<90, head); E=64 tokens.
// ---------------------------------------------------------------------------
__global__ __launch_bounds__(256)
void attn_oc_kernel(const float* __restrict__ QKV,     // [B*90*64][768]
                    const float* __restrict__ dpk_w, const float* __restrict__ dpk_b,
                    const float* __restrict__ dpv_w, const float* __restrict__ dpv_b,
                    float* __restrict__ OT,            // (B,90,8,64,32)
                    float* __restrict__ OH, int N)
{
    const int h = blockIdx.x, n = blockIdx.y, b = blockIdx.z;
    const int tid = threadIdx.x;

    __shared__ float q[64][32], k[64][32], v[64][32];
    __shared__ float qe[64][32];
    __shared__ float kt[8][32], vt[8][32], kte[8][32];
    __shared__ float sk[64][8], sv[64][8];
    __shared__ float sch[32][32];
    __shared__ float pw[96];

    const size_t tok0 = (size_t)(b * N + n) * 64;
    for (int i = tid; i < 64 * 32; i += 256) {
        int e = i >> 5, d = i & 31;
        size_t base = (tok0 + e) * 768 + h * 32 + d;
        q[e][d] = QKV[base];
        k[e][d] = QKV[base + 256];
        v[e][d] = QKV[base + 512];
    }
    if (tid < 96) pw[tid] = powf(0.7f, (float)tid);
    __syncthreads();

    // Dynamic low-rank projection weights: softmax over DPR=8.
    if (tid < 128) {
        int e = tid & 63;
        const float* wgt = (tid < 64) ? dpk_w : dpv_w;
        const float* bb  = (tid < 64) ? dpk_b : dpv_b;
        const float (*src)[32] = (tid < 64) ? k : v;
        float l[8]; float mx = -1e30f;
        for (int c = 0; c < 8; ++c) {
            float s = bb[c];
            for (int d = 0; d < 32; ++d) s += src[e][d] * wgt[c * 32 + d];
            l[c] = s; mx = fmaxf(mx, s);
        }
        float sum = 0.f;
        for (int c = 0; c < 8; ++c) { l[c] = expf(l[c] - mx); sum += l[c]; }
        float inv = 1.f / sum;
        float (*dst)[8] = (tid < 64) ? sk : sv;
        for (int c = 0; c < 8; ++c) dst[e][c] = l[c] * inv;
    }
    __syncthreads();

    // k_t[c][d] = sum_e k[e][d]*sk[e][c] ; same for v_t.
    for (int i = tid; i < 512; i += 256) {
        int which = i >> 8, c = (i >> 5) & 7, d = i & 31;
        float s = 0.f;
        if (which == 0) { for (int e = 0; e < 64; ++e) s += k[e][d] * sk[e][c]; kt[c][d] = s; }
        else            { for (int e = 0; e < 64; ++e) s += v[e][d] * sv[e][c]; vt[c][d] = s; }
    }
    __syncthreads();

    // EMA over tokens: qe (A=64), kte (A=8).
    for (int i = tid; i < 64 * 32; i += 256) {
        int g = i >> 5, d = i & 31;
        float s = 0.f;
        for (int a = 0; a <= g; ++a) s += emaE(g, a, pw) * q[a][d];
        qe[g][d] = s;
    }
    {
        int g = tid >> 5, d = tid & 31;
        float s = 0.f;
        for (int a = 0; a <= g && a < 8; ++a) s += emaE(g, a, pw) * kt[a][d];
        kte[g][d] = s;
    }
    __syncthreads();

    // Token attention: score (64x8) * sqrt(HD), softmax over 8, @ v_t.
    if (tid < 64) {
        int e = tid;
        float s[8]; float mx = -1e30f;
        for (int f = 0; f < 8; ++f) {
            float t = 0.f;
            for (int d = 0; d < 32; ++d) t += qe[e][d] * kte[f][d];
            t *= 5.656854249492381f;     // sqrt(32)
            s[f] = t; mx = fmaxf(mx, t);
        }
        float sum = 0.f;
        for (int f = 0; f < 8; ++f) { s[f] = expf(s[f] - mx); sum += s[f]; }
        float inv = 1.f / sum;
        size_t ob = ((((size_t)(b * N + n)) * 8 + h) * 64 + e) * 32;
        for (int d = 0; d < 32; ++d) {
            float o = 0.f;
            for (int f = 0; f < 8; ++f) o += s[f] * inv * vt[f][d];
            OT[ob + d] = o;
        }
    }
    // Hidden attention: score_h = q^T k * sqrt(H=64), softmax, out = v @ p^T.
    for (int i = tid; i < 32 * 32; i += 256) {
        int e = i >> 5, f = i & 31;
        float s = 0.f;
        for (int a = 0; a < 64; ++a) s += q[a][e] * k[a][f];
        sch[e][f] = s * 8.0f;
    }
    __syncthreads();
    if (tid < 32) {
        int e = tid; float mx = -1e30f;
        for (int f = 0; f < 32; ++f) mx = fmaxf(mx, sch[e][f]);
        float sum = 0.f;
        for (int f = 0; f < 32; ++f) { float t = expf(sch[e][f] - mx); sch[e][f] = t; sum += t; }
        float inv = 1.f / sum;
        for (int f = 0; f < 32; ++f) sch[e][f] *= inv;
    }
    __syncthreads();
    for (int i = tid; i < 64 * 32; i += 256) {
        int a = i >> 5, e = i & 31;
        float s = 0.f;
        for (int f = 0; f < 32; ++f) s += sch[e][f] * v[a][f];
        OH[((((size_t)(b * N + n)) * 8 + h) * 64 + a) * 32 + e] = s;
    }
}

// ---------------------------------------------------------------------------
// Over-token attention (ot): one block per (b, n<64, head); T=90 tokens.
// ---------------------------------------------------------------------------
__global__ __launch_bounds__(256)
void attn_ot_kernel(const float* __restrict__ QKV,     // [B*64*90][768]
                    float* __restrict__ OT,            // (B,64,8,90,32)
                    float* __restrict__ OH)
{
    const int h = blockIdx.x, n = blockIdx.y, b = blockIdx.z;
    const int tid = threadIdx.x;
    const int T = 90;

    __shared__ float Aq[90][32], Ak[90][32], Av[90][32];
    __shared__ float Tm[90][32];
    __shared__ float Sh[32][32];
    __shared__ float Sc[8][90];
    __shared__ float pw[96];

    const size_t tok0 = (size_t)(b * 64 + n) * 90;
    for (int i = tid; i < 90 * 32; i += 256) {
        int e = i >> 5, d = i & 31;
        size_t base = (tok0 + e) * 768 + h * 32 + d;
        Aq[e][d] = QKV[base];
        Ak[e][d] = QKV[base + 256];
        Av[e][d] = QKV[base + 512];
    }
    if (tid < 96) pw[tid] = powf(0.7f, (float)tid);
    __syncthreads();

    // Hidden attention on raw q,k (scale sqrt(H=90)).
    for (int i = tid; i < 32 * 32; i += 256) {
        int e = i >> 5, f = i & 31;
        float s = 0.f;
        for (int a = 0; a < 90; ++a) s += Aq[a][e] * Ak[a][f];
        Sh[e][f] = s * 9.486832980505138f;
    }
    __syncthreads();
    if (tid < 32) {
        int e = tid; float mx = -1e30f;
        for (int f = 0; f < 32; ++f) mx = fmaxf(mx, Sh[e][f]);
        float sum = 0.f;
        for (int f = 0; f < 32; ++f) { float t = expf(Sh[e][f] - mx); Sh[e][f] = t; sum += t; }
        float inv = 1.f / sum;
        for (int f = 0; f < 32; ++f) Sh[e][f] *= inv;
    }
    __syncthreads();
    for (int i = tid; i < 90 * 32; i += 256) {
        int a = i >> 5, e = i & 31;
        float s = 0.f;
        for (int f = 0; f < 32; ++f) s += Sh[e][f] * Av[a][f];
        OH[(((size_t)(b * 64 + n) * 8 + h) * 90 + a) * 32 + e] = s;
    }
    __syncthreads();

    // q <- EMA90 @ q ; k <- EMA90 @ k  (in place via Tm)
    for (int i = tid; i < 90 * 32; i += 256) {
        int g = i >> 5, d = i & 31;
        float s = 0.f;
        for (int a = 0; a <= g; ++a) s += emaE(g, a, pw) * Aq[a][d];
        Tm[g][d] = s;
    }
    __syncthreads();
    for (int i = tid; i < 90 * 32; i += 256) { int g = i >> 5, d = i & 31; Aq[g][d] = Tm[g][d]; }
    __syncthreads();
    for (int i = tid; i < 90 * 32; i += 256) {
        int g = i >> 5, d = i & 31;
        float s = 0.f;
        for (int a = 0; a <= g; ++a) s += emaE(g, a, pw) * Ak[a][d];
        Tm[g][d] = s;
    }
    __syncthreads();
    for (int i = tid; i < 90 * 32; i += 256) { int g = i >> 5, d = i & 31; Ak[g][d] = Tm[g][d]; }
    __syncthreads();

    // Token attention (90x90) in row chunks of 8.
    for (int e0 = 0; e0 < 90; e0 += 8) {
        int rows = (90 - e0 < 8) ? (90 - e0) : 8;
        for (int i = tid; i < rows * 90; i += 256) {
            int r = i / 90, f = i % 90;
            float s = 0.f;
            for (int d = 0; d < 32; ++d) s += Aq[e0 + r][d] * Ak[f][d];
            Sc[r][f] = s * 5.656854249492381f;   // sqrt(HD=32)
        }
        __syncthreads();
        if (tid < rows) {
            int r = tid; float mx = -1e30f;
            for (int f = 0; f < 90; ++f) mx = fmaxf(mx, Sc[r][f]);
            float sum = 0.f;
            for (int f = 0; f < 90; ++f) { float t = expf(Sc[r][f] - mx); Sc[r][f] = t; sum += t; }
            float inv = 1.f / sum;
            for (int f = 0; f < 90; ++f) Sc[r][f] *= inv;
        }
        __syncthreads();
        for (int i = tid; i < rows * 32; i += 256) {
            int r = i >> 5, d = i & 31;
            float s = 0.f;
            for (int f = 0; f < 90; ++f) s += Sc[r][f] * Av[f][d];
            OT[(((size_t)(b * 64 + n) * 8 + h) * T + (e0 + r)) * 32 + d] = s;
        }
        __syncthreads();
    }
}

// CARD head merge: (BN, NH, T, HD) -> (BN, T, 256) with '(h1 h2 h3)->h2 (h3 h1)'
__global__ void merge_kernel(const float* __restrict__ o, float* __restrict__ y,
                             int BN, int T)
{
    size_t total = (size_t)BN * T * 256;
    size_t i = (size_t)blockIdx.x * 256 + threadIdx.x;
    if (i >= total) return;
    int c = (int)(i & 255);
    size_t j = i >> 8;
    int t2 = (int)(j % T);
    size_t bn = j / T;
    int m = c >> 7, g = (c >> 5) & 3, d = c & 31;
    size_t f = (size_t)g * T * 64 + (size_t)t2 * 64 + (size_t)m * 32 + d;
    int hh = (int)(f / ((size_t)T * 32));
    size_t rr = f % ((size_t)T * 32);
    int t = (int)(rr >> 5), hd = (int)(rr & 31);
    y[i] = o[(((bn * 8) + hh) * T + t) * 32 + hd];
}

// BatchNorm (training stats): per-channel mean/var over M rows of [M][C].
__global__ __launch_bounds__(256)
void bn_stats_kernel(const float* __restrict__ x, float* __restrict__ mean,
                     float* __restrict__ var, int M, int C)
{
    int c = blockIdx.x;
    float s = 0.f, s2 = 0.f;
    for (int r = threadIdx.x; r < M; r += 256) {
        float v = x[(size_t)r * C + c];
        s += v; s2 += v * v;
    }
    __shared__ float sh[256], sh2[256];
    sh[threadIdx.x] = s; sh2[threadIdx.x] = s2;
    __syncthreads();
    for (int o = 128; o > 0; o >>= 1) {
        if (threadIdx.x < o) { sh[threadIdx.x] += sh[threadIdx.x + o];
                               sh2[threadIdx.x] += sh2[threadIdx.x + o]; }
        __syncthreads();
    }
    if (threadIdx.x == 0) {
        float m = sh[0] / (float)M;
        mean[c] = m;
        var[c] = sh2[0] / (float)M - m * m;
    }
}

__global__ void bn_apply_kernel(const float* __restrict__ x, float* __restrict__ y,
                                const float* __restrict__ mean, const float* __restrict__ var,
                                const float* __restrict__ g, const float* __restrict__ b,
                                size_t n, int C)
{
    size_t i = (size_t)blockIdx.x * 256 + threadIdx.x;
    if (i >= n) return;
    int c = (int)(i % C);
    y[i] = (x[i] - mean[c]) * rsqrtf(var[c] + 1e-5f) * g[c] + b[c];
}

__global__ void add2_kernel(const float* __restrict__ a, const float* __restrict__ b,
                            float* __restrict__ y, size_t n)
{
    size_t i = (size_t)blockIdx.x * 256 + threadIdx.x;
    if (i < n) y[i] = a[i] + b[i];
}

// de-normalize + transpose to (B, PLEN=96, CIN=64)
__global__ void finalize_kernel(const float* __restrict__ zout,  // [1024][96]
                                const float* __restrict__ zm, const float* __restrict__ zs,
                                float* __restrict__ out)
{
    int i = blockIdx.x * 256 + threadIdx.x;    // 98304 total
    if (i >= 98304) return;
    int c = i & 63;
    int p = (i >> 6) % 96;
    int b = i / (96 * 64);
    int bc = b * 64 + c;
    out[i] = zout[(size_t)bc * 96 + p] * (zs[bc] + 1e-4f) + zm[bc];
}

// ---------------------------------------------------------------------------
extern "C" void kernel_launch(void* const* d_in, const int* in_sizes, int n_in,
                              void* d_out, int out_size, void* d_ws, size_t ws_size,
                              hipStream_t stream)
{
    (void)in_sizes; (void)n_in; (void)out_size; (void)ws_size;
    // Assumed leaf order: jax tree flatten (dict keys sorted): params then x.
    auto F = [&](int i) { return (const float*)d_in[i]; };

    struct AttnPtr {
        const float *dpk_b, *dpk_w, *dpv_b, *dpv_w;
        const float *ff1a_b, *ff1a_w, *ff1b_b, *ff1b_w;
        const float *ff2a_b, *ff2a_w, *ff2b_b, *ff2b_w;
        const float *na_b, *na_g, *np1_b, *np1_g, *np2_b, *np2_g;
        const float *qkv_b, *qkv_w;
    };
    auto loadAttn = [&](int base) {
        AttnPtr a;
        a.dpk_b = F(base + 0);  a.dpk_w = F(base + 1);
        a.dpv_b = F(base + 2);  a.dpv_w = F(base + 3);
        a.ff1a_b = F(base + 4); a.ff1a_w = F(base + 5);
        a.ff1b_b = F(base + 6); a.ff1b_w = F(base + 7);
        a.ff2a_b = F(base + 8); a.ff2a_w = F(base + 9);
        a.ff2b_b = F(base + 10); a.ff2b_w = F(base + 11);
        a.na_b = F(base + 12);  a.na_g = F(base + 13);
        a.np1_b = F(base + 14); a.np1_g = F(base + 15);
        a.np2_b = F(base + 16); a.np2_g = F(base + 17);
        a.qkv_b = F(base + 18); a.qkv_w = F(base + 19);
        return a;
    };
    struct LayerPtr { const float *mlp_b, *mlp_w, *norm_b, *norm_g; AttnPtr oc, ot; };

    const float* cls = F(0);
    LayerPtr L[2];
    for (int l = 0; l < 2; ++l) {
        int base = 1 + l * 44;
        L[l].mlp_b = F(base + 0); L[l].mlp_w = F(base + 1);
        L[l].norm_b = F(base + 2); L[l].norm_g = F(base + 3);
        L[l].oc = loadAttn(base + 4);
        L[l].ot = loadAttn(base + 24);
    }
    const float* pos    = F(89);
    const float* wip_b  = F(90);
    const float* wip_w  = F(91);
    const float* wout_b = F(92);
    const float* wout_w = F(93);
    const float* x      = F(94);

    // Workspace carve-up (floats).
    float* wsp = (float*)d_ws;
    size_t off = 0;
    auto alloc = [&](size_t n) { float* p = wsp + off; off += n; return p; };
    const size_t T = (size_t)16 * 64 * 90 * 256;   // 23,592,960 elems per act tensor
    const int Mtok = 92160;                        // 16*90*64 == 16*64*90

    float* inputs = alloc(T);
    float* srcT   = alloc(T);
    float* qkv    = alloc(3 * T);
    float* otok   = alloc(T);
    float* ohid   = alloc(T);
    float* m1     = alloc(T);
    float* m2     = alloc(T);
    float* hbuf   = alloc(2 * T);
    float* f1     = alloc(T);
    float* f2     = alloc(T);
    float* o1buf  = alloc(T);
    float* o2buf  = alloc(T);
    float* sumbuf = alloc(T);
    float* zm     = alloc(1024);
    float* zs     = alloc(1024);
    float* zout   = alloc(1024 * 96);
    float* bnm    = alloc(256);
    float* bnv    = alloc(256);

    const unsigned GT = (unsigned)((T + 255) / 256);

    auto gemm = [&](int act, const float* X_, const float* W_, const float* b_,
                    float* Y_, int M, int N, int K) {
        int guard = ((M % 64) || (N % 128)) ? 1 : 0;   // stores only; staging OOB-safe
        dim3 grid((N + 127) / 128, (M + 63) / 64);
        if (act)
            gemm_wmma<1><<<grid, dim3(256), 0, stream>>>(X_, W_, b_, Y_, M, N, K, guard);
        else
            gemm_wmma<0><<<grid, dim3(256), 0, stream>>>(X_, W_, b_, Y_, M, N, K, guard);
    };
    auto bnorm = [&](const float* X_, float* Y_, const float* g_, const float* b_, int M) {
        bn_stats_kernel<<<256, 256, 0, stream>>>(X_, bnm, bnv, M, 256);
        size_t n = (size_t)M * 256;
        bn_apply_kernel<<<(unsigned)((n + 255) / 256), 256, 0, stream>>>(X_, Y_, bnm, bnv, g_, b_, n, 256);
    };
    auto add2 = [&](const float* a, const float* b, float* y, size_t n) {
        add2_kernel<<<(unsigned)((n + 255) / 256), 256, 0, stream>>>(a, b, y, n);
    };

    // One full _attention block; src_/dst_ are [Mtok][256] contiguous.
    auto attention = [&](const AttnPtr& P, const float* src_, float* dst_, bool over_hidden) {
        gemm(0, src_, P.qkv_w, P.qkv_b, qkv, Mtok, 768, 256);
        if (over_hidden) {
            attn_oc_kernel<<<dim3(8, 90, 16), 256, 0, stream>>>(
                qkv, P.dpk_w, P.dpk_b, P.dpv_w, P.dpv_b, otok, ohid, 90);
            merge_kernel<<<GT, 256, 0, stream>>>(otok, m1, 16 * 90, 64);
            merge_kernel<<<GT, 256, 0, stream>>>(ohid, m2, 16 * 90, 64);
        } else {
            attn_ot_kernel<<<dim3(8, 64, 16), 256, 0, stream>>>(qkv, otok, ohid);
            merge_kernel<<<GT, 256, 0, stream>>>(otok, m1, 16 * 64, 90);
            merge_kernel<<<GT, 256, 0, stream>>>(ohid, m2, 16 * 64, 90);
        }
        bnorm(m1, m1, P.np1_g, P.np1_b, Mtok);
        bnorm(m2, m2, P.np2_g, P.np2_b, Mtok);
        gemm(1, m1, P.ff1a_w, P.ff1a_b, hbuf, Mtok, 512, 256);
        gemm(0, hbuf, P.ff1b_w, P.ff1b_b, f1, Mtok, 256, 512);
        gemm(1, m2, P.ff2a_w, P.ff2a_b, hbuf, Mtok, 512, 256);
        gemm(0, hbuf, P.ff2b_w, P.ff2b_b, f2, Mtok, 256, 512);
        add2(f1, f2, f1, T);
        add2(src_, f1, f1, T);
        bnorm(f1, dst_, P.na_g, P.na_b, Mtok);
    };

    // ---- forward ----
    rowstats_kernel<<<1024, 256, 0, stream>>>(x, zm, zs);
    embed_kernel<<<GT, 256, 0, stream>>>(x, wip_w, wip_b, pos, cls, zm, zs, inputs);

    for (int l = 0; l < 2; ++l) {
        // over-channel attention on (B, TTN, CIN, D)
        transpose_ct_kernel<<<GT, 256, 0, stream>>>(inputs, srcT, 64, 90);
        attention(L[l].oc, srcT, srcT, true);
        transpose_ct_kernel<<<GT, 256, 0, stream>>>(srcT, o1buf, 90, 64);  // -> (B,CIN,TTN,D)
        // over-token attention on (B, CIN, TTN, D)
        attention(L[l].ot, o1buf, o2buf, false);
        // out = (o1 + o2) @ mlp^T + mlp_b + inputs ; then BatchNorm
        add2(o1buf, o2buf, sumbuf, T);
        gemm(0, sumbuf, L[l].mlp_w, L[l].mlp_b, srcT, Mtok, 256, 256);
        add2(srcT, inputs, srcT, T);
        bnorm(srcT, inputs, L[l].norm_g, L[l].norm_b, Mtok);
    }

    // Output head: (B*CIN, TTN*D=23040) @ wout^T -> (B*CIN, 96); de-normalize.
    gemm(0, inputs, wout_w, wout_b, zout, 1024, 96, 23040);
    finalize_kernel<<<(98304 + 255) / 256, 256, 0, stream>>>(zout, zm, zs, (float*)d_out);
}
